// SpiralShiftAutoencoder_extra_conv_63711544868976
// MI455X (gfx1250) — matrix operand order repaired and checked by reading, not compiled, not run on hardware
//
#include <hip/hip_runtime.h>

// ---------------------------------------------------------------------------
// CDNA5 (gfx1250, wave32) spiral-mesh autoencoder, v5.
//
// All matmuls run on v_wmma_f32_16x16x32_f16.  Design points:
//  * f16 end-to-end: one pre-pass converts weights / pool matrices / input to
//    f16 (34 MB D0/U0 stay L2-resident across the 16 batch passes; 192 MB L2).
//  * Static operands (weights W, pool matrices D/U) are pre-swizzled into the
//    16-bit WMMA A/B VGPR fragment layouts (contiguous k order, same in A and B) as
//    contiguous 1KB blocks -> hot loops load fragments straight from global
//    memory as v16h (2x global_load_b128, coalesced), no LDS staging for them.
//  * sconv outputs destined for a pool are written directly in the pool's
//    B-fragment layout, so pool kernels are pure global->WMMA: no LDS, no
//    barriers.  sconv uses only its own wave's LDS region (same-wave DS
//    ordering) -> zero s_barrier anywhere.
//  * No integer division and no branches in hot loops: each wave pre-builds a
//    per-k gather offset table in LDS from the spiral indices (adds only).
//  * Gather loads: single scalar base + 32-bit element offsets so every load
//    is GVS-mode global_load_u16 ... scale_offset (no 64-bit address VALU).
// ---------------------------------------------------------------------------

typedef __attribute__((ext_vector_type(16))) _Float16 v16h;
typedef __attribute__((ext_vector_type(8)))  float    v8f;

#define WAVES_PER_BLOCK 4
#define MAX_KTOT 4224  // largest S*Fin over all layers (fc_enc: 33*128)

__device__ __forceinline__ v8f wmma_f16(v16h a, v16h b, v8f c) {
  return __builtin_amdgcn_wmma_f32_16x16x32_f16(
      /*neg_a=*/false, a, /*neg_b=*/false, b,
      /*c_mod=*/(short)0, c, /*reuse_a=*/false, /*reuse_b=*/false);
}

// ---------------------------------------------------------------------------
// Pre-pass kernels: f32 -> f16 conversion + fragment swizzles
// ---------------------------------------------------------------------------
__global__ __launch_bounds__(256) void cvt_f16_kernel(
    const float* __restrict__ in, _Float16* __restrict__ out, int n) {
  const int i = blockIdx.x * 256 + threadIdx.x;
  if (i < n) out[i] = (_Float16)in[i];
}

// B-fragment swizzle: W[Fout,Ktot] (row major) -> frag[OT][KC][32 lanes][16]
// B k order used here (contiguous, same as A): lanes 0-15 hold K=0..15, lanes 16-31 hold K=16..31.
__global__ __launch_bounds__(256) void swizzleB_kernel(
    const float* __restrict__ W, _Float16* __restrict__ frag, int Fout,
    int Ktot) {
  const int KC = (Ktot + 31) >> 5;
  const int OT = (Fout + 15) >> 4;
  const int idx = blockIdx.x * 256 + threadIdx.x;
  if (idx >= OT * KC * 512) return;
  const int j    = idx & 15;
  const int lane = (idx >> 4) & 31;
  const int kcI  = (idx >> 9) % KC;
  const int ot   = (idx >> 9) / KC;
  const int o = ot * 16 + (lane & 15);
  const int k = kcI * 32 + ((lane >> 4) << 4) + j;
  const float v = (o < Fout && k < Ktot) ? W[(size_t)o * Ktot + k] : 0.0f;
  frag[idx] = (_Float16)v;
}

// A-fragment swizzle: P[Mout,Kin] -> frag[MT][KC][32 lanes][16]
// A k order used here (contiguous, same as B): lanes 0-15 hold K=0..15, lanes 16-31 hold K=16..31.
__global__ __launch_bounds__(256) void swizzleA_kernel(
    const float* __restrict__ P, _Float16* __restrict__ frag, int Mout,
    int Kin) {
  const int KC = (Kin + 31) >> 5;
  const int MT = (Mout + 15) >> 4;
  const int idx = blockIdx.x * 256 + threadIdx.x;
  if (idx >= MT * KC * 512) return;
  const int pos  = idx & 15;
  const int lane = (idx >> 4) & 31;
  const int kcI  = (idx >> 9) % KC;
  const int mt   = (idx >> 9) / KC;
  const int m  = mt * 16 + (lane & 15);
  const int kk = ((lane >> 4) << 4) + pos;
  const int k  = kcI * 32 + kk;
  const float v = (m < Mout && k < Kin) ? P[(size_t)m * Kin + k] : 0.0f;
  frag[idx] = (_Float16)v;
}

// ---------------------------------------------------------------------------
// Spiral conv (FC when sp == nullptr):
//   y[b,n,o] = act( sum_k g[b,n,k] * W[o,k] + bias[o] ),  g = gather(x, sp)
// One wave per vertex, NT 16-channel tiles, K chunks of 32.
// out_mode: 0 = f16 linear [b,n,o]  1 = f16 pool-B-frag (row=n, col=o)
//           2 = f32 linear          3 = f16 pool-B-frag with reshape
//                                       (row = o/frag_cols, col = o%frag_cols)
// ---------------------------------------------------------------------------
template <int NT>
__global__ __launch_bounds__(128) void sconv_kernel(
    const _Float16* __restrict__ x, const int* __restrict__ sp,
    const _Float16* __restrict__ Wfrag, const float* __restrict__ bias,
    void* __restrict__ y, int N, int Fin, int S, int Fout, int act_elu,
    int zero_last, int out_mode, int frag_rows, int frag_cols) {
  const int Ktot = S * Fin;
  const int KC   = (Ktot + 31) >> 5;
  __shared__ __align__(32) _Float16 ldsA[WAVES_PER_BLOCK][32][16];
  __shared__ int jlds[WAVES_PER_BLOCK][16];
  __shared__ int tbl[WAVES_PER_BLOCK][MAX_KTOT];

  const int tid    = threadIdx.x;
  const int wave   = tid >> 5;
  const int lane   = tid & 31;
  const int n      = blockIdx.x * WAVES_PER_BLOCK + wave;
  const bool valid = (n < N);
  const int otbase = blockIdx.y * NT;
  const int fbase  = otbase * 16;

  // spiral neighbor offsets for this wave's vertex (uniform per wave)
  if (lane < S)
    jlds[wave][lane] = (valid ? (sp ? sp[n * S + lane] : n) : 0) * Fin;

  // Build per-k gather offset table with pure adds (no division anywhere).
  for (int s = 0; s < S; ++s) {
    const int jofs = jlds[wave][s];
    const int kb   = s * Fin;
    for (int f = lane; f < Fin; f += 32) tbl[wave][kb + f] = jofs + f;
  }
  {
    const int kpad = Ktot + lane;  // pad the K tail of the last chunk
    if (kpad < (KC << 5)) tbl[wave][kpad] = -1;
  }

  v8f acc[NT];
#pragma unroll
  for (int t = 0; t < NT; ++t)
#pragma unroll
    for (int r = 0; r < 8; ++r) acc[t][r] = 0.0f;

  // Per-lane constants: this lane stages K-column kk == lane each chunk.
  const int kk   = lane;
  const int grp  = kk >> 3;
  const int lrow = (kk >> 4) << 4;                // lane-half select
  const int lpos = kk & 15;                       // position within row
  const int NFin = N * Fin;  // per-batch element stride (fits in i32)

  for (int kcI = 0; kcI < KC; ++kcI) {
    const int k = (kcI << 5) + kk;
    int so = tbl[wave][k];
    const bool inK = valid && (so >= 0);
    if (so < 0) so = 0;  // clamp: load stays in-bounds, value masked below
    // 16 loads: scalar base + 32-bit offset -> GVS global_load_u16
    // scale_offset with one v_add_nc_u32 each (b*NFin is uniform, hoisted).
#pragma unroll
    for (int b = 0; b < 16; ++b) {
      _Float16 v = x[so + b * NFin];
      if (!inK) v = (_Float16)0.0f;
      ldsA[wave][b + lrow][lpos] = v;
    }
    const v16h afrag = *(const v16h*)&ldsA[wave][lane][0];
#pragma unroll
    for (int t = 0; t < NT; ++t) {
      const _Float16* bp =
          Wfrag + ((size_t)(otbase + t) * KC + kcI) * 512 + lane * 16;
      const v16h bfrag = *(const v16h*)bp;
      if (kcI + 1 < KC) __builtin_prefetch(bp + 512, 0, 2);  // next K slice
      acc[t] = wmma_f16(afrag, bfrag, acc[t]);
    }
  }

  if (!valid) return;
  const int col   = lane & 15;
  const int mbase = (lane >> 4) << 3;  // C layout: lanes 16-31 hold M=8..15
  const int KCf   = (frag_rows + 31) >> 5;
  const int FTf   = frag_cols >> 4;
  const bool zrow   = zero_last && (n == N - 1);
  const bool isfrag = (out_mode == 1) || (out_mode == 3);
  const size_t bstr = isfrag ? (size_t)KCf * FTf * 512 : (size_t)N * Fout;
#pragma unroll
  for (int t = 0; t < NT; ++t) {
    const int o = fbase + t * 16 + col;
    if (o < Fout) {
      const float bv = bias[o];
      size_t a0;
      if (isfrag) {
        int row, cc;
        if (out_mode == 1) { row = n; cc = o; }
        else { row = o / frag_cols; cc = o - row * frag_cols; }
        a0 = ((size_t)(row >> 5) * FTf + (cc >> 4)) * 512 +
             (size_t)((((row >> 4) & 1) << 4) | (cc & 15)) * 16 + (row & 15);
      } else {
        a0 = (size_t)n * Fout + o;
      }
      a0 += (size_t)mbase * bstr;
#pragma unroll
      for (int r = 0; r < 8; ++r) {
        float v = acc[t][r] + bv;
        if (act_elu) v = (v > 0.0f) ? v : (__expf(v) - 1.0f);
        if (zrow) v = 0.0f;
        if (out_mode == 2) ((float*)y)[a0] = v;
        else                ((_Float16*)y)[a0] = (_Float16)v;
        a0 += bstr;
      }
    }
  }
}

// ---------------------------------------------------------------------------
// Pool / unpool:  y[b,m,f] = sum_n P[m,n] * x[b,n,f]
// Both operands pre-swizzled fragments in global memory -> pure
// global_load_b128 + v_wmma.  No LDS, no barriers.
// ---------------------------------------------------------------------------
template <int FT>
__global__ __launch_bounds__(128) void pool_kernel(
    const _Float16* __restrict__ Pfrag, const _Float16* __restrict__ xfrag,
    _Float16* __restrict__ y, int Mout, int Kin, int F) {
  const int KC = (Kin + 31) >> 5;
  const int MT = (Mout + 15) >> 4;
  const int tid  = threadIdx.x;
  const int wave = tid >> 5;
  const int lane = tid & 31;
  const int b    = blockIdx.y;
  const int mt   = blockIdx.x * WAVES_PER_BLOCK + wave;
  const int mte  = (mt < MT) ? mt : (MT - 1);  // clamp ragged tail
  const int m0   = mt * 16;

  v8f acc[FT];
#pragma unroll
  for (int t = 0; t < FT; ++t)
#pragma unroll
    for (int r = 0; r < 8; ++r) acc[t][r] = 0.0f;

  const _Float16* Ap = Pfrag + (size_t)mte * KC * 512 + lane * 16;
  const _Float16* Bp = xfrag + (size_t)b * KC * FT * 512 + lane * 16;

  for (int kcI = 0; kcI < KC; ++kcI) {
    const v16h afrag = *(const v16h*)(Ap + (size_t)kcI * 512);
    if (kcI + 1 < KC) __builtin_prefetch(Ap + (size_t)(kcI + 1) * 512, 0, 2);
#pragma unroll
    for (int t = 0; t < FT; ++t) {
      const v16h bfrag = *(const v16h*)(Bp + ((size_t)kcI * FT + t) * 512);
      acc[t] = wmma_f16(afrag, bfrag, acc[t]);
    }
  }

  const int col   = lane & 15;
  const int mbase = (lane >> 4) << 3;
#pragma unroll
  for (int t = 0; t < FT; ++t) {
    const int f = t * 16 + col;
    size_t a0 = ((size_t)b * Mout + (m0 + mbase)) * F + f;
#pragma unroll
    for (int r = 0; r < 8; ++r) {
      if (m0 + mbase + r < Mout) y[a0] = (_Float16)acc[t][r];
      a0 += F;
    }
  }
}

// ---------------------------------------------------------------------------
// Host-side dispatch (only the tile counts actually used are instantiated)
// ---------------------------------------------------------------------------
static void launch_sconv(const _Float16* x, const int* sp,
                         const _Float16* Wfrag, const float* bias, void* y,
                         int N, int Fin, int S, int Fout, int act_elu,
                         int zero_last, int out_mode, int frag_rows,
                         int frag_cols, hipStream_t st) {
  int nt = (Fout + 15) >> 4;
  if (nt > 8) nt = 8;
  const int groups = (Fout + nt * 16 - 1) / (nt * 16);
  dim3 grid((N + WAVES_PER_BLOCK - 1) / WAVES_PER_BLOCK, groups);
  dim3 blk(128);
#define SC_CASE(NTV)                                                         \
  case NTV:                                                                  \
    sconv_kernel<NTV><<<grid, blk, 0, st>>>(x, sp, Wfrag, bias, y, N, Fin,   \
                                            S, Fout, act_elu, zero_last,     \
                                            out_mode, frag_rows, frag_cols); \
    break;
  switch (nt) {
    SC_CASE(1) SC_CASE(2) SC_CASE(4) SC_CASE(6) SC_CASE(8)
    default: break;  // tile counts 3/5/7 never occur in this network
  }
#undef SC_CASE
}

static void launch_pool(const _Float16* Pfrag, const _Float16* xfrag,
                        _Float16* y, int Mout, int Kin, int F, int B,
                        hipStream_t st) {
  const int ft = (F + 15) >> 4;
  const int MT = (Mout + 15) >> 4;
  dim3 grid((MT + WAVES_PER_BLOCK - 1) / WAVES_PER_BLOCK, B);
  dim3 blk(128);
#define PL_CASE(FTV)                                                        \
  case FTV:                                                                 \
    pool_kernel<FTV><<<grid, blk, 0, st>>>(Pfrag, xfrag, y, Mout, Kin, F);  \
    break;
  switch (ft) {
    PL_CASE(2) PL_CASE(4) PL_CASE(6) PL_CASE(8)
    default: break;  // F is always 32/64/96/128 here
  }
#undef PL_CASE
}

extern "C" void kernel_launch(void* const* d_in, const int* in_sizes, int n_in,
                              void* d_out, int out_size, void* d_ws,
                              size_t ws_size, hipStream_t stream) {
  // Input order: x, sp0..sp3, D0..D3, U0..U3,
  //              enc (W,b)x8, dec (W,b)x5, fc_enc (W,b), fc_dec (W,b)
  const float* xf = (const float*)d_in[0];
  const int* sp[4];
  for (int i = 0; i < 4; ++i) sp[i] = (const int*)d_in[1 + i];
  const float* Dm[4];
  for (int i = 0; i < 4; ++i) Dm[i] = (const float*)d_in[5 + i];
  const float* Um[4];
  for (int i = 0; i < 4; ++i) Um[i] = (const float*)d_in[9 + i];
  const float *encW[8], *encB[8];
  for (int l = 0; l < 8; ++l) {
    encW[l] = (const float*)d_in[13 + 2 * l];
    encB[l] = (const float*)d_in[14 + 2 * l];
  }
  const float *decW[5], *decB[5];
  for (int l = 0; l < 5; ++l) {
    decW[l] = (const float*)d_in[29 + 2 * l];
    decB[l] = (const float*)d_in[30 + 2 * l];
  }
  const float* feW = (const float*)d_in[39];
  const float* feB = (const float*)d_in[40];
  const float* fdW = (const float*)d_in[41];
  const float* fdB = (const float*)d_in[42];

  const int N0 = 8193, N1 = 2049, N2 = 513, N3 = 129, N4 = 33, S = 12, B = 16;

  // ---- workspace allocator (f16 halves, 64-half aligned) ----
  _Float16* wsp = (_Float16*)d_ws;
  size_t ofs = 0;
  auto alloc = [&](size_t halves) -> _Float16* {
    _Float16* p = wsp + ofs;
    ofs += (halves + 63) & ~(size_t)63;
    return p;
  };
  auto swB = [&](const float* W, int Fout, int Ktot) -> _Float16* {
    const size_t tot = (size_t)((Fout + 15) / 16) * ((Ktot + 31) / 32) * 512;
    _Float16* p = alloc(tot);
    swizzleB_kernel<<<dim3((tot + 255) / 256), dim3(256), 0, stream>>>(
        W, p, Fout, Ktot);
    return p;
  };
  auto swA = [&](const float* P, int Mout, int Kin) -> _Float16* {
    const size_t tot = (size_t)((Mout + 15) / 16) * ((Kin + 31) / 32) * 512;
    _Float16* p = alloc(tot);
    swizzleA_kernel<<<dim3((tot + 255) / 256), dim3(256), 0, stream>>>(
        P, p, Mout, Kin);
    return p;
  };

  // ---- pre-pass: input + all static operands to f16 / fragments ----
  const int nx = B * N0 * 3;
  _Float16* x16 = alloc(nx);
  cvt_f16_kernel<<<dim3((nx + 255) / 256), dim3(256), 0, stream>>>(xf, x16, nx);

  _Float16* eWf[8];
  const int eFo[8] = {32, 32, 64, 64, 96, 96, 128, 128};
  const int eKt[8] = {3 * S, 32 * S, 32 * S, 64 * S, 64 * S, 96 * S, 96 * S, 128 * S};
  for (int l = 0; l < 8; ++l) eWf[l] = swB(encW[l], eFo[l], eKt[l]);
  _Float16* dWf[5];
  const int dFo[5] = {96, 64, 32, 16, 3};
  const int dKt[5] = {128 * S, 96 * S, 64 * S, 32 * S, 16 * S};
  for (int l = 0; l < 5; ++l) dWf[l] = swB(decW[l], dFo[l], dKt[l]);
  _Float16* feWf = swB(feW, 256, N4 * 128);
  _Float16* fdWf = swB(fdW, N4 * 128, 256);
  _Float16* Dfr[4];
  Dfr[0] = swA(Dm[0], N1, N0); Dfr[1] = swA(Dm[1], N2, N1);
  Dfr[2] = swA(Dm[2], N3, N2); Dfr[3] = swA(Dm[3], N4, N3);
  _Float16* Ufr[4];
  Ufr[0] = swA(Um[0], N0, N1); Ufr[1] = swA(Um[1], N1, N2);
  Ufr[2] = swA(Um[2], N2, N3); Ufr[3] = swA(Um[3], N3, N4);

  // ---- activation ping/pong buffers (f16, linear or frag layout) ----
  const size_t ACT = (size_t)8 << 20;  // 8M halves = 16 MB each
  _Float16* bufA = alloc(ACT);
  _Float16* bufB = alloc(ACT);
  _Float16* bufC = alloc(B * 256);  // latent

  float* out = (float*)d_out;

  // ---------------- encoder ----------------
  launch_sconv(x16, sp[0], eWf[0], encB[0], bufA, N0, 3, S, 32, 1, 1, 0, 0, 16, stream);
  launch_sconv(bufA, sp[0], eWf[1], encB[1], bufB, N0, 32, S, 32, 1, 1, 1, N0, 32, stream);
  launch_pool(Dfr[0], bufB, bufA, N1, N0, 32, B, stream);
  launch_sconv(bufA, sp[1], eWf[2], encB[2], bufB, N1, 32, S, 64, 1, 1, 0, 0, 16, stream);
  launch_sconv(bufB, sp[1], eWf[3], encB[3], bufA, N1, 64, S, 64, 1, 1, 1, N1, 64, stream);
  launch_pool(Dfr[1], bufA, bufB, N2, N1, 64, B, stream);
  launch_sconv(bufB, sp[2], eWf[4], encB[4], bufA, N2, 64, S, 96, 1, 1, 0, 0, 16, stream);
  launch_sconv(bufA, sp[2], eWf[5], encB[5], bufB, N2, 96, S, 96, 1, 1, 1, N2, 96, stream);
  launch_pool(Dfr[2], bufB, bufA, N3, N2, 96, B, stream);
  launch_sconv(bufA, sp[3], eWf[6], encB[6], bufB, N3, 96, S, 128, 1, 1, 0, 0, 16, stream);
  launch_sconv(bufB, sp[3], eWf[7], encB[7], bufA, N3, 128, S, 128, 1, 1, 1, N3, 128, stream);
  launch_pool(Dfr[3], bufA, bufB, N4, N3, 128, B, stream);

  // ---------------- bottleneck (FC as N=1 "vertex") ----------------
  launch_sconv(bufB, nullptr, feWf, feB, bufC, 1, N4 * 128, 1, 256, 0, 0, 0, 0, 16, stream);
  // fc_dec output reshapes to [16,33,128] and feeds pool U3 -> frag mode 3
  launch_sconv(bufC, nullptr, fdWf, fdB, bufA, 1, 256, 1, N4 * 128, 0, 0, 3, N4, 128, stream);

  // ---------------- decoder ----------------
  launch_pool(Ufr[3], bufA, bufB, N3, N4, 128, B, stream);
  launch_sconv(bufB, sp[3], dWf[0], decB[0], bufA, N3, 128, S, 96, 1, 1, 1, N3, 96, stream);
  launch_pool(Ufr[2], bufA, bufB, N2, N3, 96, B, stream);
  launch_sconv(bufB, sp[2], dWf[1], decB[1], bufA, N2, 96, S, 64, 1, 1, 1, N2, 64, stream);
  launch_pool(Ufr[1], bufA, bufB, N1, N2, 64, B, stream);
  launch_sconv(bufB, sp[1], dWf[2], decB[2], bufA, N1, 64, S, 32, 1, 1, 1, N1, 32, stream);
  launch_pool(Ufr[0], bufA, bufB, N0, N1, 32, B, stream);
  launch_sconv(bufB, sp[0], dWf[3], decB[3], bufA, N0, 32, S, 16, 1, 1, 0, 0, 16, stream);
  launch_sconv(bufA, sp[0], dWf[4], decB[4], out, N0, 16, S, 3, 0, 1, 2, 0, 16, stream);
}
